// SigGraInferNet_GCN_75849122448214
// MI455X (gfx1250) — matrix operand order repaired and checked by reading, not compiled, not run on hardware
//
#include <hip/hip_runtime.h>
#include <hip/hip_bf16.h>
#include <math.h>

// ---------------------------------------------------------------------------
// Types for WMMA bf16 (gfx1250, wave32)
// ---------------------------------------------------------------------------
typedef __bf16 bf16_t;
typedef bf16_t bfx16 __attribute__((ext_vector_type(16)));
typedef bf16_t bfx8  __attribute__((ext_vector_type(8)));
typedef float  fx8   __attribute__((ext_vector_type(8)));

// Model dims
static constexpr int BS    = 8;
static constexpr int NN    = 2048;
static constexpr int PPI   = 512;
static constexpr int H     = 256;
static constexpr int FIN   = 1024;
static constexpr int NRES  = 3;
static constexpr long long ROWS = (long long)BS * NN;          // 16384
static constexpr float EPS = 1e-5f;

#if defined(__gfx1250__) && __has_builtin(__builtin_amdgcn_global_load_async_to_lds_b128) && __has_builtin(__builtin_amdgcn_s_wait_asynccnt)
#define USE_ASYNC_LDS 1
#else
#define USE_ASYNC_LDS 0
#endif

#if USE_ASYNC_LDS
typedef int v4i_vs __attribute__((vector_size(16)));
typedef __attribute__((address_space(1))) v4i_vs* as1_v4i;
typedef __attribute__((address_space(3))) v4i_vs* as3_v4i;
#endif

__device__ __forceinline__ unsigned short f2bf(float f) {
  unsigned int u = __float_as_uint(f);
  u += 0x7fffu + ((u >> 16) & 1u);     // round-to-nearest-even
  return (unsigned short)(u >> 16);
}

// ---------------------------------------------------------------------------
// f32 -> bf16 conversion (grid-stride, layout preserved)
// ---------------------------------------------------------------------------
__global__ __launch_bounds__(256) void k_f32_to_bf16(
    const float* __restrict__ in, unsigned short* __restrict__ out, long long n) {
  long long i = (long long)blockIdx.x * blockDim.x + threadIdx.x;
  long long stride = (long long)gridDim.x * blockDim.x;
  for (; i < n; i += stride) out[i] = f2bf(in[i]);
}

// ---------------------------------------------------------------------------
// f32 -> bf16 with transpose: out[z][n][m] = bf16(in[z][m][n])
// in: [Mrows, Ncols] f32; out: [Ncols, Mrows] bf16. Tiles 32x32 via LDS.
// grid: (Ncols/32, Mrows/32, batch)
// ---------------------------------------------------------------------------
__global__ __launch_bounds__(256) void k_transpose_bf16(
    const float* __restrict__ in, unsigned short* __restrict__ out,
    int Mrows, int Ncols, long long sIn, long long sOut) {
  __shared__ float tile[32][33];
  const float*    ip = in  + (long long)blockIdx.z * sIn;
  unsigned short* op = out + (long long)blockIdx.z * sOut;
  int n0 = blockIdx.x * 32;
  int m0 = blockIdx.y * 32;
  int tx = threadIdx.x & 31;
  int ty = threadIdx.x >> 5;   // 0..7
  for (int r = ty; r < 32; r += 8)
    tile[r][tx] = ip[(long long)(m0 + r) * Ncols + n0 + tx];
  __syncthreads();
  for (int r = ty; r < 32; r += 8)
    op[(long long)(n0 + r) * Mrows + m0 + tx] = f2bf(tile[tx][r]);
}

// ---------------------------------------------------------------------------
// bf16 GEMM:  C[M,N] = A[M,K] * B[K,N]  with B supplied TRANSPOSED:
//   A  : [M,K] bf16 row-major
//   BT : [N,K] bf16 row-major  (= B column-major)
//   C  : [M,N] f32 row-major
// block: 256 threads (8 waves), 128x128 C tile, K-tile 64
// grid (N/128, M/128, batch). M%128==0, N%128==0, K%64==0 assumed.
// ---------------------------------------------------------------------------
#define KT 64
#define LS 72    // LDS row stride in bf16 elems (64 + 8 pad); 144B, 16B-aligned

__global__ __launch_bounds__(256) void k_gemm_bf16(
    const bf16_t* __restrict__ Ag, const bf16_t* __restrict__ BTg,
    float* __restrict__ Cg, int M, int N, int K,
    long long sA, long long sB, long long sC) {
  __shared__ __align__(16) bf16_t ldsA[128 * LS];
  __shared__ __align__(16) bf16_t ldsB[128 * LS];

  const int tid = threadIdx.x;
  const int z   = blockIdx.z;
  const bf16_t* A  = Ag  + z * sA;
  const bf16_t* BT = BTg + z * sB;
  float*        C  = Cg  + z * sC;

  const int nBase = blockIdx.x * 128;
  const int mBase = blockIdx.y * 128;

  const int wave  = tid >> 5;
  const int lane  = tid & 31;
  const int l15   = lane & 15;
  const int lhi   = lane >> 4;          // half-wave select
  const int waveM = (wave & 3) * 32;    // 4 waves along M
  const int waveN = (wave >> 2) * 64;   // 2 waves along N

  fx8 acc[2][4] = {};

  // stagers: each thread copies 32 contiguous bf16 (4 x b128) of one row
  const int row = tid >> 1;             // 0..127
  const int col = (tid & 1) * 32;       // 0 or 32
  const bf16_t* ap = A  + (long long)(mBase + row) * K + col;
  const bf16_t* bp = BT + (long long)(nBase + row) * K + col;
  bf16_t* la = &ldsA[row * LS + col];
  bf16_t* lb = &ldsB[row * LS + col];

  for (int kk = 0; kk < K; kk += KT) {
    const bf16_t* apk = ap + kk;
    const bf16_t* bpk = bp + kk;
#if USE_ASYNC_LDS
#pragma unroll
    for (int u = 0; u < 4; ++u) {
      __builtin_amdgcn_global_load_async_to_lds_b128(
          (as1_v4i)(v4i_vs*)(void*)(apk + u * 8),
          (as3_v4i)(v4i_vs*)(void*)(la + u * 8), 0, 0);
      __builtin_amdgcn_global_load_async_to_lds_b128(
          (as1_v4i)(v4i_vs*)(void*)(bpk + u * 8),
          (as3_v4i)(v4i_vs*)(void*)(lb + u * 8), 0, 0);
    }
    __builtin_amdgcn_s_wait_asynccnt(0);
#else
#pragma unroll
    for (int u = 0; u < 4; ++u) {
      bfx8 av = *(const bfx8*)(apk + u * 8);
      bfx8 bv = *(const bfx8*)(bpk + u * 8);
      *(bfx8*)(la + u * 8) = av;
      *(bfx8*)(lb + u * 8) = bv;
    }
#endif
    if (kk + KT < K) {                 // prefetch next K-slice
      __builtin_prefetch((const void*)(apk + KT), 0, 1);
      __builtin_prefetch((const void*)(bpk + KT), 0, 1);
    }
    __syncthreads();

#pragma unroll
    for (int ks = 0; ks < KT; ks += 32) {
      // A fragments: lanes 0-15 hold K {0..7,16..23}, lanes 16-31 {8..15,24..31}
      bfx16 afr[2];
#pragma unroll
      for (int mt = 0; mt < 2; ++mt) {
        int m  = waveM + mt * 16 + l15;
        int k0 = ks + lhi * 8;
        bfx8 lo = *(const bfx8*)&ldsA[m * LS + k0];
        bfx8 hi = *(const bfx8*)&ldsA[m * LS + k0 + 16];
        afr[mt] = __builtin_shufflevector(lo, hi, 0,1,2,3,4,5,6,7,8,9,10,11,12,13,14,15);
      }
      // B fragments: lanes 0-15 hold K {0..7,16..23}, lanes 16-31 {8..15,24..31}
      bfx16 bfr[4];
#pragma unroll
      for (int nt = 0; nt < 4; ++nt) {
        int n  = waveN + nt * 16 + l15;
        int k0 = ks + lhi * 8;
        bfx8 lo = *(const bfx8*)&ldsB[n * LS + k0];
        bfx8 hi = *(const bfx8*)&ldsB[n * LS + k0 + 16];
        bfr[nt] = __builtin_shufflevector(lo, hi, 0,1,2,3,4,5,6,7,8,9,10,11,12,13,14,15);
      }
#pragma unroll
      for (int mt = 0; mt < 2; ++mt)
#pragma unroll
        for (int nt = 0; nt < 4; ++nt)
          acc[mt][nt] = __builtin_amdgcn_wmma_f32_16x16x32_bf16(
              false, afr[mt], false, bfr[nt], (short)0, acc[mt][nt], false, false);
    }
    __syncthreads();
  }

  // C layout: VGPR r -> M = r + 8*lhi, N = lane&15
#pragma unroll
  for (int mt = 0; mt < 2; ++mt)
#pragma unroll
    for (int nt = 0; nt < 4; ++nt)
#pragma unroll
      for (int r = 0; r < 8; ++r) {
        int m = mBase + waveM + mt * 16 + lhi * 8 + r;
        int n = nBase + waveN + nt * 16 + l15;
        C[(long long)m * N + n] = acc[mt][nt][r];
      }
}

// ---------------------------------------------------------------------------
// BatchNorm (train-mode batch stats) helpers
// ---------------------------------------------------------------------------
__global__ void k_bn_zero(float* stats) { stats[threadIdx.x] = 0.f; }  // 1x512

__global__ __launch_bounds__(256) void k_bn_accum(
    const float* __restrict__ h, float* __restrict__ sum, float* __restrict__ sumsq) {
  int c  = threadIdx.x;             // channel
  int r0 = blockIdx.x * 64;
  float s = 0.f, s2 = 0.f;
  for (int r = r0; r < r0 + 64; ++r) {
    float v = h[(long long)r * H + c];
    s += v; s2 += v * v;
  }
  atomicAdd(&sum[c], s);
  atomicAdd(&sumsq[c], s2);
}

__global__ void k_bn_finalize(const float* __restrict__ sum, const float* __restrict__ sumsq,
                              const float* __restrict__ gamma,
                              float* __restrict__ mu, float* __restrict__ scale) {
  int c = threadIdx.x;
  float m = sum[c] / (float)ROWS;
  float v = sumsq[c] / (float)ROWS - m * m;
  mu[c]    = m;
  scale[c] = gamma[c] * rsqrtf(v + EPS);
}

// y = relu((h-mu)*scale + beta) [+ skip]; writes f32 G and bf16 copy
__global__ __launch_bounds__(256) void k_bn_apply(
    const float* __restrict__ h, const float* __restrict__ mu,
    const float* __restrict__ scale, const float* __restrict__ beta,
    const float* __restrict__ skip, float* __restrict__ g,
    unsigned short* __restrict__ gbf, long long n) {
  long long i = (long long)blockIdx.x * blockDim.x + threadIdx.x;
  long long stride = (long long)gridDim.x * blockDim.x;
  for (; i < n; i += stride) {
    int c = (int)(i & (H - 1));
    float v = (h[i] - mu[c]) * scale[c] + beta[c];
    v = fmaxf(v, 0.f);
    if (skip) v += skip[i];
    g[i]   = v;
    gbf[i] = f2bf(v);
  }
}

// ---------------------------------------------------------------------------
// GGE: relu(relu(a@w1+b1)@w2+b2)  — a:[8,1024]  -> h:[8,256]
// ---------------------------------------------------------------------------
__global__ __launch_bounds__(256) void k_gge(
    const float* __restrict__ a, const float* __restrict__ w1, const float* __restrict__ b1,
    const float* __restrict__ w2, const float* __restrict__ b2, float* __restrict__ hout) {
  __shared__ float av[FIN];
  __shared__ float t1[128];
  int b = blockIdx.x, t = threadIdx.x;
  for (int k = t; k < FIN; k += 256) av[k] = a[(long long)b * FIN + k];
  __syncthreads();
  if (t < 128) {
    float acc = b1[t];
    for (int k = 0; k < FIN; ++k) acc += av[k] * w1[k * 128 + t];
    t1[t] = fmaxf(acc, 0.f);
  }
  __syncthreads();
  float acc = b2[t];
  for (int k = 0; k < 128; ++k) acc += t1[k] * w2[k * H + t];
  hout[(long long)b * H + t] = fmaxf(acc, 0.f);
}

// ---------------------------------------------------------------------------
// GAGA attention: att = softmax_n(G.h); e = sum_n att_n * G[n,:]
// ---------------------------------------------------------------------------
__global__ __launch_bounds__(256) void k_gaga(
    const float* __restrict__ G, const float* __restrict__ h, float* __restrict__ e) {
  __shared__ float att[NN];
  __shared__ float red[256];
  __shared__ float hv[H];
  int b = blockIdx.x, t = threadIdx.x;
  hv[t] = h[(long long)b * H + t];
  __syncthreads();
  const float* Gb = G + (long long)b * NN * H;
  for (int n = t; n < NN; n += 256) {
    const float* row = Gb + (long long)n * H;
    float d = 0.f;
    for (int j = 0; j < H; ++j) d += row[j] * hv[j];
    att[n] = d;
  }
  __syncthreads();
  float mx = -INFINITY;
  for (int n = t; n < NN; n += 256) mx = fmaxf(mx, att[n]);
  red[t] = mx; __syncthreads();
  for (int s = 128; s > 0; s >>= 1) { if (t < s) red[t] = fmaxf(red[t], red[t + s]); __syncthreads(); }
  mx = red[0]; __syncthreads();
  float ls = 0.f;
  for (int n = t; n < NN; n += 256) { float ex = expf(att[n] - mx); att[n] = ex; ls += ex; }
  red[t] = ls; __syncthreads();
  for (int s = 128; s > 0; s >>= 1) { if (t < s) red[t] += red[t + s]; __syncthreads(); }
  float inv = 1.f / red[0];
  float acc = 0.f;
  for (int n = 0; n < NN; ++n) acc += Gb[(long long)n * H + t] * att[n];
  e[(long long)b * H + t] = acc * inv;
}

// ---------------------------------------------------------------------------
// LED head + log_softmax — per batch
// ---------------------------------------------------------------------------
__global__ __launch_bounds__(256) void k_led(
    const float* __restrict__ ha, const float* __restrict__ eA,
    const float* __restrict__ hb, const float* __restrict__ eB,
    const float* __restrict__ cw, const float* __restrict__ cb,
    const float* __restrict__ p1w, const float* __restrict__ p1b,
    const float* __restrict__ p2w, const float* __restrict__ p2b,
    float* __restrict__ out) {
  __shared__ float sea[2 * H], seb[2 * H], cnn[2 * H], diff[2 * H], x[H];
  int b = blockIdx.x, t = threadIdx.x;
  sea[t]     = ha[(long long)b * H + t];
  sea[H + t] = eA[(long long)b * H + t];
  seb[t]     = hb[(long long)b * H + t];
  seb[H + t] = eB[(long long)b * H + t];
  __syncthreads();
  for (int o = t; o < 2 * H; o += 256) {
    float a0 = cb[o], a1 = cb[o];
    const float* w = cw + (long long)o * (2 * H);
    for (int c = 0; c < 2 * H; ++c) { a0 += w[c] * sea[c]; a1 += w[c] * seb[c]; }
    cnn[o] = fmaxf(a0, a1);
  }
  for (int c = t; c < 2 * H; c += 256) diff[c] = sea[c] - seb[c];
  __syncthreads();
  {
    float acc = p1b[t];
    for (int k = 0; k < 2 * H; ++k) acc += cnn[k]  * p1w[k * H + t];
    for (int k = 0; k < 2 * H; ++k) acc += diff[k] * p1w[(2 * H + k) * H + t];
    x[t] = fmaxf(acc, 0.f);
  }
  __syncthreads();
  if (t == 0) {
    float l0 = p2b[0], l1 = p2b[1];
    for (int k = 0; k < H; ++k) { l0 += x[k] * p2w[k * 2]; l1 += x[k] * p2w[k * 2 + 1]; }
    float m = fmaxf(l0, l1);
    float lse = m + logf(expf(l0 - m) + expf(l1 - m));
    out[b * 2]     = l0 - lse;
    out[b * 2 + 1] = l1 - lse;
  }
}

// ---------------------------------------------------------------------------
// Host orchestration
// ---------------------------------------------------------------------------
extern "C" void kernel_launch(void* const* d_in, const int* in_sizes, int n_in,
                              void* d_out, int out_size, void* d_ws, size_t ws_size,
                              hipStream_t stream) {
  (void)in_sizes; (void)n_in; (void)out_size; (void)ws_size;
  const float* a_in   = (const float*)d_in[0];
  const float* bio_a  = (const float*)d_in[1];
  const float* adjA   = (const float*)d_in[2];
  const float* b_in   = (const float*)d_in[3];
  const float* bio_b  = (const float*)d_in[4];
  const float* adjB   = (const float*)d_in[5];
  const float* W0     = (const float*)d_in[6];
  const float* gamma0 = (const float*)d_in[7];
  const float* beta0  = (const float*)d_in[8];
  const float* W_res  = (const float*)d_in[9];
  const float* g_res  = (const float*)d_in[10];
  const float* b_res  = (const float*)d_in[11];
  const float* g1_w   = (const float*)d_in[12];
  const float* g1_b   = (const float*)d_in[13];
  const float* g2_w   = (const float*)d_in[14];
  const float* g2_b   = (const float*)d_in[15];
  const float* conv_w = (const float*)d_in[16];
  const float* conv_b = (const float*)d_in[17];
  const float* p1_w   = (const float*)d_in[18];
  const float* p1_b   = (const float*)d_in[19];
  const float* p2_w   = (const float*)d_in[20];
  const float* p2_b   = (const float*)d_in[21];
  float* out = (float*)d_out;

  // ---- workspace carve-out (256B aligned) ----
  size_t off = 0;
  char* base = (char*)d_ws;
  auto carve = [&](size_t bytes) -> void* {
    void* p = base + off;
    off = (off + bytes + 255) & ~(size_t)255;
    return p;
  };
  unsigned short* adjBf = (unsigned short*)carve((size_t)BS * NN * NN * 2);   // 64 MB
  unsigned short* bioBf = (unsigned short*)carve((size_t)ROWS * PPI * 2);     // 16 MB
  unsigned short* W0T   = (unsigned short*)carve((size_t)PPI * H * 2);        // [256][512]
  unsigned short* WrT   = (unsigned short*)carve((size_t)NRES * H * H * 2);   // [3][256][256]
  float*          Tf32  = (float*)carve((size_t)ROWS * H * 4);                // 16 MB
  unsigned short* TbfT  = (unsigned short*)carve((size_t)ROWS * H * 2);       // [8][256][2048]
  float*          Hraw  = (float*)carve((size_t)ROWS * H * 4);                // 16 MB
  unsigned short* Gbf   = (unsigned short*)carve((size_t)ROWS * H * 2);       // 8 MB
  float*          GA    = (float*)carve((size_t)ROWS * H * 4);                // 16 MB
  float*          GB    = (float*)carve((size_t)ROWS * H * 4);                // 16 MB
  float*          stats = (float*)carve(4 * H * 4);   // [sum|sumsq|mu|scale]
  float* sumv = stats, *sumsq = stats + H, *mu = stats + 2 * H, *scale = stats + 3 * H;
  float* ha = (float*)carve((size_t)BS * H * 4);
  float* hb = (float*)carve((size_t)BS * H * 4);
  float* eA = (float*)carve((size_t)BS * H * 4);
  float* eB = (float*)carve((size_t)BS * H * 4);

  // ---- weight transpose+convert (once per launch) ----
  {
    dim3 g(H / 32, PPI / 32, 1);
    k_transpose_bf16<<<g, 256, 0, stream>>>(W0, W0T, PPI, H, 0, 0);
  }
  {
    dim3 g(H / 32, H / 32, NRES);
    k_transpose_bf16<<<g, 256, 0, stream>>>(W_res, WrT, H, H,
                                            (long long)H * H, (long long)H * H);
  }

  auto run_bn = [&](const float* gamma, const float* beta, const float* skip, float* Gout) {
    k_bn_zero<<<1, 512, 0, stream>>>(stats);
    k_bn_accum<<<ROWS / 64, 256, 0, stream>>>(Hraw, sumv, sumsq);
    k_bn_finalize<<<1, H, 0, stream>>>(sumv, sumsq, gamma, mu, scale);
    k_bn_apply<<<4096, 256, 0, stream>>>(Hraw, mu, scale, beta, skip, Gout,
                                         Gbf, ROWS * (long long)H);
  };

  auto adj_gemm = [&]() {   // Hraw[z] = adjBf[z] @ T[z]  (BT = TbfT[z])
    dim3 g(H / 128, NN / 128, BS);
    k_gemm_bf16<<<g, 256, 0, stream>>>((const bf16_t*)adjBf, (const bf16_t*)TbfT,
                                       Hraw, NN, H, NN,
                                       (long long)NN * NN, (long long)H * NN,
                                       (long long)NN * H);
  };

  auto t_transpose = [&]() {   // Tf32 [8][2048][256] -> TbfT [8][256][2048]
    dim3 g(H / 32, NN / 32, BS);
    k_transpose_bf16<<<g, 256, 0, stream>>>(Tf32, TbfT, NN, H,
                                            (long long)NN * H, (long long)NN * H);
  };

  auto run_graph = [&](const float* bio, const float* adj, float* Gout) {
    k_f32_to_bf16<<<4096, 256, 0, stream>>>(bio, bioBf, (long long)ROWS * PPI);
    k_f32_to_bf16<<<8192, 256, 0, stream>>>(adj, adjBf, (long long)BS * NN * NN);
    // layer 0: T = bio @ W0  (M=16384, K=512, N=256)
    {
      dim3 g(H / 128, ROWS / 128, 1);
      k_gemm_bf16<<<g, 256, 0, stream>>>((const bf16_t*)bioBf, (const bf16_t*)W0T,
                                         Tf32, (int)ROWS, H, PPI, 0, 0, 0);
    }
    t_transpose();
    adj_gemm();
    run_bn(gamma0, beta0, nullptr, Gout);
    // residual stack
    for (int i = 0; i < NRES; ++i) {
      {
        dim3 g(H / 128, ROWS / 128, 1);
        k_gemm_bf16<<<g, 256, 0, stream>>>((const bf16_t*)Gbf,
                                           (const bf16_t*)(WrT + (size_t)i * H * H),
                                           Tf32, (int)ROWS, H, H, 0, 0, 0);
      }
      t_transpose();
      adj_gemm();
      run_bn(g_res + (size_t)i * H, b_res + (size_t)i * H, Gout, Gout);
    }
  };

  run_graph(bio_a, adjA, GA);
  run_graph(bio_b, adjB, GB);

  k_gge<<<BS, 256, 0, stream>>>(a_in, g1_w, g1_b, g2_w, g2_b, ha);
  k_gge<<<BS, 256, 0, stream>>>(b_in, g1_w, g1_b, g2_w, g2_b, hb);
  k_gaga<<<BS, 256, 0, stream>>>(GA, ha, eA);
  k_gaga<<<BS, 256, 0, stream>>>(GB, hb, eB);
  k_led<<<BS, 256, 0, stream>>>(ha, eA, hb, eB, conv_w, conv_b,
                                p1_w, p1_b, p2_w, p2_b, out);
}
